// PointSamplingNetRadius_14637248545009
// MI455X (gfx1250) — matrix operand order repaired and checked by reading, not compiled, not run on hardware
//
#include <hip/hip_runtime.h>
#include <hip/hip_bf16.h>

typedef __attribute__((ext_vector_type(16))) __bf16 v16bf;
typedef __attribute__((ext_vector_type(8)))  float  v8f;

#define BB 16
#define MM 16384
#define NCH3 512
#define NEG_INF (-3.4e38f)

// ---------------- small utility kernels ----------------

__global__ void zero_kernel(float* p, int n) {
    int i = blockIdx.x * 256 + threadIdx.x;
    if (i < n) p[i] = 0.f;
}

// Pack W (O x K fp32, row-major) into WMMA A-fragment layout, bf16:
// packed[((ot*KT + kt)*32 + lane)*16 + j]
//   row = lane&15, hi = lane>>4
//   k_local = j + (hi ? 16 : 0)
__global__ void pack_kernel(const float* __restrict__ w2, const float* __restrict__ w3,
                            __bf16* __restrict__ p2, __bf16* __restrict__ p3) {
    int idx = blockIdx.x * 256 + threadIdx.x;
    if (idx < 131072) {                 // w3: O=512,K=256 -> 32 ot * 8 kt * 32 * 16
        int j = idx & 15, lane = (idx >> 4) & 31, kt = (idx >> 9) & 7, ot = idx >> 12;
        int hi = lane >> 4, row = lane & 15;
        int kl = j + (hi ? 16 : 0);
        int o = ot * 16 + row, k = kt * 32 + kl;
        p3[idx] = (__bf16)w3[o * 256 + k];
    }
    int idx2 = idx - 131072;
    if (idx2 >= 0 && idx2 < 16384) {    // w2: O=256,K=64 -> 16 ot * 2 kt * 32 * 16
        int j = idx2 & 15, lane = (idx2 >> 4) & 31, kt = (idx2 >> 9) & 1, ot = idx2 >> 10;
        int hi = lane >> 4, row = lane & 15;
        int kl = j + (hi ? 16 : 0);
        int o = ot * 16 + row, k = kt * 32 + kl;
        p2[idx2] = (__bf16)w2[o * 64 + k];
    }
}

// BN finalize: a = g*rsqrt(var+eps), d = be - mu*a
__global__ void finalize_kernel(const float* __restrict__ S, const float* __restrict__ Q,
                                const float* __restrict__ g, const float* __restrict__ be,
                                float* __restrict__ a, float* __restrict__ d, int C) {
    int c = blockIdx.x * 256 + threadIdx.x;
    if (c < C) {
        const float invN = 1.f / ((float)BB * (float)MM);
        float mu  = S[c] * invN;
        float var = fmaxf(Q[c] * invN - mu * mu, 0.f);
        float aa  = g[c] * rsqrtf(var + 1e-5f);
        a[c] = aa;
        d[c] = fmaf(-mu, aa, be[c]);
    }
}

// ---------------- layer 0: 3 -> 32 (scalar) + stats ----------------

__global__ void conv0_kernel(const float* __restrict__ coord,
                             const float* __restrict__ w0, const float* __restrict__ b0,
                             __bf16* __restrict__ y0, float* gS, float* gQ) {
    __shared__ float wl[96], bl[32], sS[32], sQ[32];
    int t = threadIdx.x;
    if (t < 96) wl[t] = w0[t];
    if (t < 32) { bl[t] = b0[t]; sS[t] = 0.f; sQ[t] = 0.f; }
    __syncthreads();
    int b = blockIdx.y;
    int m = blockIdx.x * 256 + t;
    size_t ci = ((size_t)b * MM + m) * 3;
    float cx = coord[ci], cy = coord[ci + 1], cz = coord[ci + 2];
    int lane = t & 31;
    for (int c = 0; c < 32; ++c) {
        float v = fmaf(wl[c * 3], cx, fmaf(wl[c * 3 + 1], cy, fmaf(wl[c * 3 + 2], cz, bl[c])));
        y0[((size_t)b * 32 + c) * MM + m] = (__bf16)v;
        float s = v, q = v * v;
        #pragma unroll
        for (int off = 16; off; off >>= 1) { s += __shfl_xor(s, off); q += __shfl_xor(q, off); }
        if (lane == 0) { atomicAdd(&sS[c], s); atomicAdd(&sQ[c], q); }
    }
    __syncthreads();
    if (t < 32) { atomicAdd(&gS[t], sS[t]); atomicAdd(&gQ[t], sQ[t]); }
}

// ---------------- layer 1: 32 -> 64 (scalar, BN0+ReLU fused on input) ----------------

__global__ void conv1_kernel(const __bf16* __restrict__ y0,
                             const float* __restrict__ w1, const float* __restrict__ b1,
                             const float* __restrict__ a0, const float* __restrict__ d0,
                             __bf16* __restrict__ y1, float* gS, float* gQ) {
    __shared__ float wl[64 * 32], al[32], dl[32], sS[64], sQ[64];
    int t = threadIdx.x;
    for (int i = t; i < 2048; i += 256) wl[i] = w1[i];
    if (t < 32) { al[t] = a0[t]; dl[t] = d0[t]; }
    if (t < 64) { sS[t] = 0.f; sQ[t] = 0.f; }
    __syncthreads();
    int b = blockIdx.y;
    int m = blockIdx.x * 256 + t;
    int lane = t & 31;
    float x[32];
    #pragma unroll
    for (int c = 0; c < 32; ++c) {
        float v = (float)y0[((size_t)b * 32 + c) * MM + m];
        x[c] = fmaxf(fmaf(al[c], v, dl[c]), 0.f);
    }
    for (int o = 0; o < 64; ++o) {
        float acc = b1[o];
        #pragma unroll
        for (int c = 0; c < 32; ++c) acc = fmaf(wl[o * 32 + c], x[c], acc);
        y1[((size_t)b * 64 + o) * MM + m] = (__bf16)acc;
        float s = acc, q = acc * acc;
        #pragma unroll
        for (int off = 16; off; off >>= 1) { s += __shfl_xor(s, off); q += __shfl_xor(q, off); }
        if (lane == 0) { atomicAdd(&sS[o], s); atomicAdd(&sQ[o], q); }
    }
    __syncthreads();
    if (t < 64) { atomicAdd(&gS[t], sS[t]); atomicAdd(&gQ[t], sQ[t]); }
}

// ---------------- WMMA GEMM: Y(OxMt) = W(OxK) * relu(bn(X(KxMt))) ----------------
// X: bf16 [B][K][M]. Wp: packed A-fragments. 512 threads = 16 waves, Mt=64.
// Each wave owns ONE 16-row o-tile per rep (4 m-tiles): acc = 32 VGPRs -> no spills.
// FINAL=false: write bf16 + accumulate BN stats.  FINAL=true: write fp16 logits (+bias).

template<int K, int O, bool FINAL>
__global__ void gemm_kernel(const __bf16* __restrict__ X, const __bf16* __restrict__ Wp,
                            const float* __restrict__ scl, const float* __restrict__ sh,
                            const float* __restrict__ bias,
                            __bf16* __restrict__ Ybf, _Float16* __restrict__ Yh,
                            float* gS, float* gQ) {
    constexpr int Mt = 64;
    constexpr int KT = K / 32;
    constexpr int STRIDE = K + 16;              // elements; 32B-aligned rows
    extern __shared__ char smem[];
    float* aL = (float*)smem;                   // K
    float* dL = aL + K;                         // K
    float* sS = dL + K;                         // O (only if !FINAL)
    float* sQ = sS + (FINAL ? 0 : O);
    __bf16* Xl = (__bf16*)(sQ + (FINAL ? 0 : O));

    int t = threadIdx.x;
    int b = blockIdx.y;
    int m0 = blockIdx.x * Mt;

    for (int i = t; i < K; i += 512) { aL[i] = scl[i]; dL[i] = sh[i]; }
    if constexpr (!FINAL)
        for (int i = t; i < O; i += 512) { sS[i] = 0.f; sQ[i] = 0.f; }
    __syncthreads();

    // stage X tile transposed into LDS: Xl[m][k], BN+ReLU applied
    for (int l = t; l < K * Mt; l += 512) {
        int k = l / Mt, m = l % Mt;             // consecutive t -> consecutive m (coalesced)
        float v = (float)X[((size_t)b * K + k) * MM + m0 + m];
        v = fmaxf(fmaf(aL[k], v, dL[k]), 0.f);
        Xl[m * STRIDE + k] = (__bf16)v;
    }
    __syncthreads();

    int wave = t >> 5, lane = t & 31;
    int mloc = lane & 15, hi = lane >> 4;
    constexpr int REPS = O / 256;

    for (int rep = 0; rep < REPS; ++rep) {
        int ot = rep * 16 + wave;
        v8f zero = {0.f, 0.f, 0.f, 0.f, 0.f, 0.f, 0.f, 0.f};
        v8f acc[4];
        #pragma unroll
        for (int j = 0; j < 4; ++j) acc[j] = zero;

        for (int kt = 0; kt < KT; ++kt) {
            v16bf a = *(const v16bf*)(Wp + ((size_t)(ot * KT + kt) * 32 + lane) * 16);
            #pragma unroll
            for (int j = 0; j < 4; ++j) {
                v16bf bb = *(const v16bf*)(&Xl[(j * 16 + mloc) * STRIDE + kt * 32 + hi * 16]);
                acc[j] = __builtin_amdgcn_wmma_f32_16x16x32_bf16(
                    false, a, false, bb, (short)0, acc[j], false, false);
            }
        }

        #pragma unroll
        for (int j = 0; j < 4; ++j) {
            #pragma unroll
            for (int r = 0; r < 8; ++r) {
                int o = ot * 16 + r + hi * 8;   // C layout: lanes<16 rows 0..7, lanes>=16 rows 8..15
                int m = j * 16 + mloc;
                float v = acc[j][r] + bias[o];
                if constexpr (FINAL) {
                    Yh[((size_t)b * O + o) * MM + m0 + m] = (_Float16)v;
                } else {
                    Ybf[((size_t)b * O + o) * MM + m0 + m] = (__bf16)v;
                    atomicAdd(&sS[o], v);
                    atomicAdd(&sQ[o], v * v);
                }
            }
        }
    }

    if constexpr (!FINAL) {
        __syncthreads();
        for (int i = t; i < O; i += 512) { atomicAdd(&gS[i], sS[i]); atomicAdd(&gQ[i], sQ[i]); }
    }
}

// ---------------- per-point logsumexp over 512 channels ----------------

__global__ void lse_kernel(const _Float16* __restrict__ logits, float* __restrict__ lse) {
    int b = blockIdx.y;
    int m = blockIdx.x * 256 + threadIdx.x;
    const _Float16* p = logits + (size_t)b * NCH3 * MM + m;
    float mx = NEG_INF, s = 0.f;
    for (int c = 0; c < NCH3; ++c) {
        float v = (float)p[(size_t)c * MM];
        float nm = fmaxf(mx, v);
        s = fmaf(s, __expf(mx - nm), __expf(v - nm));
        mx = nm;
    }
    lse[(size_t)b * MM + m] = mx + __logf(s);
}

// ---------------- top-32 per (b,s) row + radius mask ----------------
// logQ = logit - lse; 32 rounds of "next lexicographic max below last pick"
// (exactly reproduces stable argsort(-Q): descending value, ascending index).

__global__ void topk_kernel(const _Float16* __restrict__ logits, const float* __restrict__ lse,
                            const float* __restrict__ coord,
                            int* __restrict__ outS, int* __restrict__ outG) {
    __shared__ float redV[256];
    __shared__ int   redI[256];
    __shared__ int   topI[32];
    int t = threadIdx.x;
    int s = blockIdx.x, b = blockIdx.y;
    const _Float16* row  = logits + ((size_t)b * NCH3 + s) * MM;
    const float*    lrow = lse + (size_t)b * MM;

    float qr[64];
    #pragma unroll
    for (int i = 0; i < 64; ++i) {
        int m = i * 256 + t;
        qr[i] = (float)row[m] - lrow[m];
    }

    float lastV = 3.4e38f; int lastI = -1;
    for (int r = 0; r < 32; ++r) {
        float bv = NEG_INF; int bi = 0x7fffffff;
        #pragma unroll
        for (int i = 0; i < 64; ++i) {
            int m = i * 256 + t;
            float v = qr[i];
            bool cand = (v < lastV) || (v == lastV && m > lastI);
            if (cand && (v > bv || (v == bv && m < bi))) { bv = v; bi = m; }
        }
        redV[t] = bv; redI[t] = bi;
        __syncthreads();
        for (int off = 128; off > 0; off >>= 1) {
            if (t < off) {
                float ov = redV[t + off]; int oi = redI[t + off];
                if (ov > redV[t] || (ov == redV[t] && oi < redI[t])) { redV[t] = ov; redI[t] = oi; }
            }
            __syncthreads();
        }
        lastV = redV[0]; lastI = redI[0];
        if (t == 0) topI[r] = lastI;
        __syncthreads();
    }

    if (t < 32) {
        int gi = topI[t], si = topI[0];
        size_t gb = ((size_t)b * MM + gi) * 3, sb = ((size_t)b * MM + si) * 3;
        float dx = coord[gb] - coord[sb];
        float dy = coord[gb + 1] - coord[sb + 1];
        float dz = coord[gb + 2] - coord[sb + 2];
        float d2 = dx * dx + dy * dy + dz * dz;
        int g = (d2 > 1.0f) ? si : gi;
        outG[((size_t)b * NCH3 + s) * 32 + t] = g;
        if (t == 0) outS[(size_t)b * NCH3 + s] = si;
    }
}

// ---------------- host launch ----------------

extern "C" void kernel_launch(void* const* d_in, const int* in_sizes, int n_in,
                              void* d_out, int out_size, void* d_ws, size_t ws_size,
                              hipStream_t stream) {
    const float* coord = (const float*)d_in[0];
    const float* w0 = (const float*)d_in[1];  const float* b0 = (const float*)d_in[2];
    const float* g0 = (const float*)d_in[3];  const float* be0 = (const float*)d_in[4];
    const float* w1 = (const float*)d_in[5];  const float* b1 = (const float*)d_in[6];
    const float* g1 = (const float*)d_in[7];  const float* be1 = (const float*)d_in[8];
    const float* w2 = (const float*)d_in[9];  const float* b2 = (const float*)d_in[10];
    const float* g2 = (const float*)d_in[11]; const float* be2 = (const float*)d_in[12];
    const float* w3 = (const float*)d_in[13]; const float* b3 = (const float*)d_in[14];

    char* ws = (char*)d_ws;
    size_t off = 0;
    __bf16* y0 = (__bf16*)(ws + off);      off += (size_t)BB * 32 * MM * 2;    // 16 MB
    __bf16* y1 = (__bf16*)(ws + off);      off += (size_t)BB * 64 * MM * 2;    // 32 MB
    __bf16* y2 = (__bf16*)(ws + off);      off += (size_t)BB * 256 * MM * 2;   // 128 MB
    _Float16* logits = (_Float16*)(ws + off); off += (size_t)BB * 512 * MM * 2; // 256 MB
    float* lse = (float*)(ws + off);       off += (size_t)BB * MM * 4;         // 1 MB
    __bf16* pw2 = (__bf16*)(ws + off);     off += 16384 * 2;
    __bf16* pw3 = (__bf16*)(ws + off);     off += 131072 * 2;
    float* stats = (float*)(ws + off);
    float* S0 = stats;        float* Q0 = S0 + 32;
    float* S1 = Q0 + 32;      float* Q1 = S1 + 64;
    float* S2 = Q1 + 64;      float* Q2 = S2 + 256;
    float* a0 = Q2 + 256;     float* d0 = a0 + 32;
    float* a1 = d0 + 32;      float* d1 = a1 + 64;
    float* a2 = d1 + 64;      float* d2 = a2 + 256;

    int* outS = (int*)d_out;
    int* outG = outS + BB * NCH3;

    zero_kernel<<<1, 256, 0, stream>>>(stats, 704);
    pack_kernel<<<576, 256, 0, stream>>>(w2, w3, pw2, pw3);

    conv0_kernel<<<dim3(MM / 256, BB), 256, 0, stream>>>(coord, w0, b0, y0, S0, Q0);
    finalize_kernel<<<1, 256, 0, stream>>>(S0, Q0, g0, be0, a0, d0, 32);

    conv1_kernel<<<dim3(MM / 256, BB), 256, 0, stream>>>(y0, w1, b1, a0, d0, y1, S1, Q1);
    finalize_kernel<<<1, 256, 0, stream>>>(S1, Q1, g1, be1, a1, d1, 64);

    // layer 2: 64 -> 256 via WMMA; LDS = (2*64 + 2*256)*4 + 64*(64+16)*2 = 12800 B
    gemm_kernel<64, 256, false><<<dim3(MM / 64, BB), 512, 12800, stream>>>(
        y1, pw2, a1, d1, b2, y2, nullptr, S2, Q2);
    finalize_kernel<<<1, 256, 0, stream>>>(S2, Q2, g2, be2, a2, d2, 256);

    // layer 3: 256 -> 512 via WMMA; LDS = 2*256*4 + 64*(256+16)*2 = 36864 B
    gemm_kernel<256, 512, true><<<dim3(MM / 64, BB), 512, 36864, stream>>>(
        y2, pw3, a2, d2, b3, nullptr, logits, nullptr, nullptr);

    lse_kernel<<<dim3(MM / 256, BB), 256, 0, stream>>>(logits, lse);
    topk_kernel<<<dim3(NCH3, BB), 256, 0, stream>>>(logits, lse, coord, outS, outG);
}
